// CosineAttention_48576080118367
// MI455X (gfx1250) — matrix operand order repaired and checked by reading, not compiled, not run on hardware
//
#include <hip/hip_runtime.h>
#include <stdint.h>

// Problem constants (match reference)
#define B_   2
#define T_   2048
#define DM_  1024
#define H_   16
#define D_   64
#define MTOK (B_ * T_)   // 4096 token rows

typedef __attribute__((ext_vector_type(16))) __bf16 v16bf;
typedef __attribute__((ext_vector_type(8)))  float  v8f;

union Frag { v16bf v; uint32_t u[8]; };

__device__ __forceinline__ uint16_t f2bf(float f) {
  uint32_t u = __float_as_uint(f);
  u += 0x7FFFu + ((u >> 16) & 1u);          // round-to-nearest-even
  return (uint16_t)(u >> 16);
}
__device__ __forceinline__ float bf2f(uint32_t bits16) {
  return __uint_as_float(bits16 << 16);
}

// ---------------------------------------------------------------------------
// fp32 -> bf16 conversion (x and weights), one thread per element
// ---------------------------------------------------------------------------
__global__ __launch_bounds__(256) void cvt_f32_bf16(const float* __restrict__ src,
                                                    uint16_t* __restrict__ dst, int n) {
  int i = blockIdx.x * 256 + threadIdx.x;
  if (i < n) dst[i] = f2bf(src[i]);
}

// ---------------------------------------------------------------------------
// bf16 GEMM:  C[m,n] = sum_k A[m,k] * W[n,k]  + bias[n]
// One wave computes a 16x64 output strip: one shared A-fragment feeds four
// back-to-back v_wmma_f32_16x16x32_bf16 per 32-wide K step (4x A reuse).
// mode 0: store bf16 head-major  [B,H,T,D]   (Q / K)
// mode 1: store bf16 transposed  [B,H,D,T]   (V)
// mode 2: store fp32 row-major   [MTOK,DM]   (output projection)
// ---------------------------------------------------------------------------
__global__ __launch_bounds__(256) void gemm16(const uint16_t* __restrict__ A,
                                              const uint16_t* __restrict__ W,
                                              const float* __restrict__ bias,
                                              void* __restrict__ dst, int mode) {
  const int lane = threadIdx.x & 31;
  const int wave = threadIdx.x >> 5;
  const int gid  = blockIdx.x * 8 + wave;       // 4096 strips total (exact)
  const int NS   = DM_ / 64;                    // 16 strips per row of tiles
  const int tm   = gid / NS;
  const int tn   = gid % NS;
  const int half = lane >> 4;
  const int lr   = lane & 15;

  const uint16_t* arow  = A + (size_t)(tm * 16 + lr) * DM_;  // A: lane holds row M=lr
  const uint16_t* wrow0 = W + (size_t)(tn * 64 + lr) * DM_;  // B: lane holds col N=lr

  v8f acc[4] = {};
  for (int k0 = 0; k0 < DM_; k0 += 32) {
    __builtin_prefetch(arow + k0 + 256, 0, 3);               // WGP-scope prefetch
    Frag a;
#pragma unroll
    for (int p = 0; p < 8; ++p) {
      const int ka = 2 * p + 16 * half;                      // contiguous k = 16*half + s
      a.u[p] = *(const uint32_t*)(arow + k0 + ka);
    }
#pragma unroll
    for (int j = 0; j < 4; ++j) {
      const uint16_t* wr = wrow0 + (size_t)(16 * j) * DM_ + k0 + 16 * half;
      __builtin_prefetch(wr + 256, 0, 3);
      Frag b;
#pragma unroll
      for (int p = 0; p < 8; ++p) b.u[p] = *(const uint32_t*)(wr + 2 * p);
      acc[j] = __builtin_amdgcn_wmma_f32_16x16x32_bf16(false, a.v, false, b.v,
                                                       (short)0, acc[j], false, false);
    }
  }

#pragma unroll
  for (int j = 0; j < 4; ++j) {
    const int n  = tn * 64 + 16 * j + lr;       // this lane's output column
    const float bn = bias[n];
#pragma unroll
    for (int r = 0; r < 8; ++r) {
      const int m = tm * 16 + r + 8 * half;     // C layout: VGPR r -> rows r / r+8
      const float val = acc[j][r] + bn;
      if (mode == 2) {
        ((float*)dst)[(size_t)m * DM_ + n] = val;
      } else {
        const int b = m >> 11, t = m & (T_ - 1);  // T_ = 2048
        const int h = n >> 6,  d = n & 63;        // D_ = 64
        if (mode == 0)
          ((uint16_t*)dst)[(size_t)((b * H_ + h) * T_ + t) * D_ + d] = f2bf(val);
        else
          ((uint16_t*)dst)[(size_t)((b * H_ + h) * D_ + d) * T_ + t] = f2bf(val);
      }
    }
  }
}

// ---------------------------------------------------------------------------
// Flash-style cosine attention. One wave owns one (b,h, 16-query) tile and
// streams 32 keys per iteration:
//   S = Q(16x64) . K^T          -> 4 wmma  (two 16x16 tiles, K-dim 64)
//   scale by tau/(|q||k|), causal mask, online softmax (shfl_xor row reduce)
//   P repacked C->A layout via per-wave LDS staging
//   O += P(16x32) . V(32x64)    -> 4 wmma
// ---------------------------------------------------------------------------
__global__ __launch_bounds__(256) void attn_fwd(const uint16_t* __restrict__ qh,
                                                const uint16_t* __restrict__ kh,
                                                const uint16_t* __restrict__ vT,
                                                const float* __restrict__ tau,
                                                uint16_t* __restrict__ ao) {
  __shared__ uint32_t ldsP[8 * 16 * 16];        // 8 waves x (16x32 bf16) = 8 KiB
  const int lane = threadIdx.x & 31;
  const int wave = threadIdx.x >> 5;
  uint16_t* pbuf = (uint16_t*)(ldsP + wave * 256);

  const int gid = blockIdx.x * 8 + wave;        // 4096 q-tiles total (exact)
  const int qt  = gid & (T_ / 16 - 1);          // 128 q-tiles per head
  const int bh  = gid >> 7;                     // b*H + h
  const int h   = bh & (H_ - 1);
  const int q0  = qt * 16;
  const int half = lane >> 4;
  const int lr   = lane & 15;

  // ---- load Q tile as two A-fragments, accumulate per-row sum of squares
  Frag qa[2];
  const uint16_t* qrow = qh + (size_t)(bh * T_ + q0 + lr) * D_;
  float ss = 0.f;
#pragma unroll
  for (int f = 0; f < 2; ++f)
#pragma unroll
    for (int p = 0; p < 8; ++p) {
      const int ka = 2 * p + 16 * half + 32 * f;
      const uint32_t u = *(const uint32_t*)(qrow + ka);
      qa[f].u[p] = u;
      const float lo = bf2f(u & 0xffffu), hi = bf2f(u >> 16);
      ss += lo * lo + hi * hi;
    }
  ss += __shfl_xor(ss, 16, 32);                         // row lr total over 64 dims
  const float qs = tau[h] / fmaxf(sqrtf(ss), 1e-12f);   // tau / max(|q|, eps)

  // broadcast per-row q scale into C-layout row order (row = r + 8*half)
  float qsr[8];
#pragma unroll
  for (int r = 0; r < 8; ++r) qsr[r] = __shfl(qs, r + 8 * half, 32);

  float m_run[8], l_run[8];
  v8f o[4] = {};
#pragma unroll
  for (int r = 0; r < 8; ++r) { m_run[r] = -1e30f; l_run[r] = 0.f; }

  const uint16_t* kbase = kh + (size_t)bh * T_ * D_;
  const uint16_t* vbase = vT + (size_t)bh * D_ * T_;

  for (int n0 = 0; n0 < q0 + 16; n0 += 32) {
    __builtin_prefetch(kbase + (size_t)(n0 + 32 + lr) * D_, 0, 3);

    // ---- K fragments (B layout: lane = key token, contiguous d pairs) + norms
    Frag kb[2][2];
    float ksc[2];
#pragma unroll
    for (int t2 = 0; t2 < 2; ++t2) {
      const uint16_t* krow = kbase + (size_t)(n0 + 16 * t2 + lr) * D_;
      float ks = 0.f;
#pragma unroll
      for (int f = 0; f < 2; ++f)
#pragma unroll
        for (int p = 0; p < 8; ++p) {
          const uint32_t u = *(const uint32_t*)(krow + 16 * half + 2 * p + 32 * f);
          kb[t2][f].u[p] = u;
          const float lo = bf2f(u & 0xffffu), hi = bf2f(u >> 16);
          ks += lo * lo + hi * hi;
        }
      ks += __shfl_xor(ks, 16, 32);
      ksc[t2] = 1.f / fmaxf(sqrtf(ks), 1e-12f);
    }

    // ---- S = Q . K^T  (two 16x16 tiles, K-dim 64)
    v8f s0 = {}, s1 = {};
    s0 = __builtin_amdgcn_wmma_f32_16x16x32_bf16(false, qa[0].v, false, kb[0][0].v, (short)0, s0, false, false);
    s0 = __builtin_amdgcn_wmma_f32_16x16x32_bf16(false, qa[1].v, false, kb[0][1].v, (short)0, s0, false, false);
    s1 = __builtin_amdgcn_wmma_f32_16x16x32_bf16(false, qa[0].v, false, kb[1][0].v, (short)0, s1, false, false);
    s1 = __builtin_amdgcn_wmma_f32_16x16x32_bf16(false, qa[1].v, false, kb[1][1].v, (short)0, s1, false, false);

    // ---- scale, causal mask, online softmax update
    float p0[8], p1[8];
#pragma unroll
    for (int r = 0; r < 8; ++r) {
      const int qg = q0 + r + 8 * half;
      float v0 = s0[r] * qsr[r] * ksc[0];
      float v1 = s1[r] * qsr[r] * ksc[1];
      if (n0 + lr      > qg) v0 = -1e30f;
      if (n0 + 16 + lr > qg) v1 = -1e30f;

      float mx = fmaxf(v0, v1);
      mx = fmaxf(mx, __shfl_xor(mx, 1, 32));
      mx = fmaxf(mx, __shfl_xor(mx, 2, 32));
      mx = fmaxf(mx, __shfl_xor(mx, 4, 32));
      mx = fmaxf(mx, __shfl_xor(mx, 8, 32));

      const float mo = m_run[r];
      const float mn = fmaxf(mo, mx);
      const float alpha = __expf(mo - mn);
      const float e0 = __expf(v0 - mn);
      const float e1 = __expf(v1 - mn);
      float rs = e0 + e1;
      rs += __shfl_xor(rs, 1, 32);
      rs += __shfl_xor(rs, 2, 32);
      rs += __shfl_xor(rs, 4, 32);
      rs += __shfl_xor(rs, 8, 32);
      l_run[r] = l_run[r] * alpha + rs;
      m_run[r] = mn;
#pragma unroll
      for (int j = 0; j < 4; ++j) o[j][r] *= alpha;
      p0[r] = e0;
      p1[r] = e1;
    }

    // ---- repack P (C layout) -> A layout via per-wave LDS (same-wave DS order)
#pragma unroll
    for (int r = 0; r < 8; ++r) {
      pbuf[(r + 8 * half) * 32 + lr]      = f2bf(p0[r]);
      pbuf[(r + 8 * half) * 32 + 16 + lr] = f2bf(p1[r]);
    }
    Frag pa;
#pragma unroll
    for (int p = 0; p < 8; ++p) {
      const int ka = 2 * p + 16 * half;
      pa.u[p] = *(const uint32_t*)(pbuf + lr * 32 + ka);
    }

    // ---- O += P . V   (V transposed: lane's d column has contiguous tokens)
#pragma unroll
    for (int j = 0; j < 4; ++j) {
      Frag vb;
      const uint16_t* vrow = vbase + (size_t)(16 * j + lr) * T_ + n0 + 16 * half;
#pragma unroll
      for (int p = 0; p < 8; ++p) vb.u[p] = *(const uint32_t*)(vrow + 2 * p);
      o[j] = __builtin_amdgcn_wmma_f32_16x16x32_bf16(false, pa.v, false, vb.v,
                                                     (short)0, o[j], false, false);
    }
  }

  // ---- epilogue: normalize by l, store bf16 to [B,T,DM]
  const int b = bh >> 4;
#pragma unroll
  for (int r = 0; r < 8; ++r) {
    const float inv = 1.f / l_run[r];
    const int t = q0 + r + 8 * half;
    uint16_t* orow = ao + (size_t)(b * T_ + t) * DM_ + h * D_;
#pragma unroll
    for (int j = 0; j < 4; ++j) orow[16 * j + lr] = f2bf(o[j][r] * inv);
  }
}

// ---------------------------------------------------------------------------
extern "C" void kernel_launch(void* const* d_in, const int* in_sizes, int n_in,
                              void* d_out, int out_size, void* d_ws, size_t ws_size,
                              hipStream_t stream) {
  const float* x   = (const float*)d_in[0];
  // d_in[1] = mask (unused: causal mask computed analytically)
  const float* Wq  = (const float*)d_in[2];
  const float* bq  = (const float*)d_in[3];
  const float* Wk  = (const float*)d_in[4];
  const float* bk  = (const float*)d_in[5];
  const float* Wv  = (const float*)d_in[6];
  const float* bv  = (const float*)d_in[7];
  const float* Wo  = (const float*)d_in[8];
  const float* bo  = (const float*)d_in[9];
  const float* tau = (const float*)d_in[10];

  // workspace layout (bf16): xb | Wq Wk Wv Wo | Qhead | Khead | V^T | attn_out
  uint16_t* xb  = (uint16_t*)d_ws;
  uint16_t* wqb = xb  + (size_t)MTOK * DM_;
  uint16_t* wkb = wqb + (size_t)DM_ * DM_;
  uint16_t* wvb = wkb + (size_t)DM_ * DM_;
  uint16_t* wob = wvb + (size_t)DM_ * DM_;
  uint16_t* qh  = wob + (size_t)DM_ * DM_;
  uint16_t* kh  = qh  + (size_t)B_ * H_ * T_ * D_;
  uint16_t* vT  = kh  + (size_t)B_ * H_ * T_ * D_;
  uint16_t* ao  = vT  + (size_t)B_ * H_ * T_ * D_;

  const int nX = MTOK * DM_;      // 4,194,304
  const int nW = DM_ * DM_;       // 1,048,576
  cvt_f32_bf16<<<nX / 256, 256, 0, stream>>>(x,  xb,  nX);
  cvt_f32_bf16<<<nW / 256, 256, 0, stream>>>(Wq, wqb, nW);
  cvt_f32_bf16<<<nW / 256, 256, 0, stream>>>(Wk, wkb, nW);
  cvt_f32_bf16<<<nW / 256, 256, 0, stream>>>(Wv, wvb, nW);
  cvt_f32_bf16<<<nW / 256, 256, 0, stream>>>(Wo, wob, nW);

  const int gemmBlocks = (MTOK / 16) * (DM_ / 64) / 8;   // 512 (16x64 strips)
  gemm16<<<gemmBlocks, 256, 0, stream>>>(xb, wqb, bq, qh, 0);   // Q head-major
  gemm16<<<gemmBlocks, 256, 0, stream>>>(xb, wkb, bk, kh, 0);   // K head-major
  gemm16<<<gemmBlocks, 256, 0, stream>>>(xb, wvb, bv, vT, 1);   // V transposed

  attn_fwd<<<(B_ * H_ * (T_ / 16)) / 8, 256, 0, stream>>>(qh, kh, vT, tau, ao);

  gemm16<<<gemmBlocks, 256, 0, stream>>>(ao, wob, bo, d_out, 2); // out proj, fp32
}
